// AnomalyAttention_26611617366118
// MI455X (gfx1250) — matrix operand order repaired and checked by reading, not compiled, not run on hardware
//
#include <hip/hip_runtime.h>
#include <cstdint>
#include <cstddef>

// AnomalyAttention for MI455X (gfx1250): bf16 WMMA GEMMs with async-to-LDS
// double-buffered staging + bandwidth-lean elementwise kernels.
// B=8, N=1024, D=1024.

typedef unsigned short u16;
typedef __attribute__((ext_vector_type(16))) __bf16 v16bf;
typedef __attribute__((ext_vector_type(8)))  float  v8f;

#define BB 8
#define NN 1024
#define DD 1024

__device__ __forceinline__ u16 f32_to_bf16_rne(float f) {
  union { float f; uint32_t u; } cv; cv.f = f;
  uint32_t u = cv.u;
  return (u16)((u + 0x7FFFu + ((u >> 16) & 1u)) >> 16);
}

struct Frag32 { uint4 lo, hi; };  // 32 bytes == v16bf

// --- CDNA5 async memory->LDS (ASYNCcnt) ------------------------------------
__device__ __forceinline__ void async_load_b128(uint32_t lds_off, const void* g) {
  // GLOBAL_LOAD_ASYNC_TO_LDS_B128: per-lane LDS[lds_off..+15] = MEM[g..+15]
  asm volatile("global_load_async_to_lds_b128 %0, %1, off"
               :: "v"(lds_off), "v"(g) : "memory");
}

#if __has_builtin(__builtin_amdgcn_s_wait_asynccnt)
#define WAIT_ASYNCCNT(n) __builtin_amdgcn_s_wait_asynccnt(n)
#else
#define WAIT_ASYNCCNT(n) asm volatile("s_wait_asynccnt %0" :: "i"(n) : "memory")
#endif

__device__ __forceinline__ uint32_t lds_offset_of(const void* p) {
  // generic LDS address = {SHARED_BASE[63:32], lds_offset[31:0]}
  return (uint32_t)(uintptr_t)p;
}

// ---------------------------------------------------------------------------
// bf16 WMMA GEMM:  C[b] = scale * A[b] (MxK, row-major) * Bt[b]^T
// Bt is stored "B-transposed": Bt[n][k], so both A and B fragments are
// contiguous b128 loads from LDS per the CDNA5 16-bit WMMA VGPR layouts.
// Block tile 128x128x32, 8 waves (4x2), wave tile 32x64 = 2x4 fragments.
// Steady-state loop is unconditional (tail peeled): barrier -> async-issue
// next tile -> s_wait_asynccnt 4 -> barrier -> 8 WMMAs from LDS.
// OUTMODE: 0 = f32 row-major, 1 = bf16 row-major, 2 = bf16 transposed (ldcT)
// ---------------------------------------------------------------------------
#define TM 128
#define TN 128
#define TK 32
#define LDSP 40   // padded row stride (u16), multiple of 8 -> 16B alignment

template <int OUTMODE>
__global__ __launch_bounds__(256)
void gemm_bf16_wmma(const u16* __restrict__ A, const u16* __restrict__ Bt,
                    void* __restrict__ Cv,
                    int M, int Nout, int Kdim,
                    long strideA, long strideB, long strideC,
                    float scale, int ldcT) {
  (void)M;
  const int b = blockIdx.z;
  const u16* Ab = A + (long)b * strideA;
  const u16* Bb = Bt + (long)b * strideB;

  __shared__ __align__(16) u16 shA[2][TM * LDSP];
  __shared__ __align__(16) u16 shB[2][TN * LDSP];

  const int tid  = threadIdx.x;
  const int lane = tid & 31;
  const int wave = tid >> 5;
  const int wr   = wave & 3;   // 4 row groups of 32
  const int wc   = wave >> 2;  // 2 col groups of 64
  const int rowBase = blockIdx.y * TM;
  const int colBase = blockIdx.x * TN;

  v8f acc[2][4] = {};

  // cooperative staging: thread covers one 16-col half of one tile row
  const int sr = tid >> 1;           // 0..127 (row within A and Bt tiles)
  const int sc = (tid & 1) * 16;     // 0 or 16 (col base; 2 x b128 = 16 u16)

  uint32_t ldsA[2], ldsB[2];
#pragma unroll
  for (int s = 0; s < 2; ++s) {
    ldsA[s] = lds_offset_of(&shA[s][sr * LDSP + sc]);
    ldsB[s] = lds_offset_of(&shB[s][sr * LDSP + sc]);
  }
  const u16* pa = Ab + (long)(rowBase + sr) * Kdim + sc;
  const u16* pb = Bb + (long)(colBase + sr) * Kdim + sc;

  const int h  = lane >> 4;          // fragment half select
  const int ml = lane & 15;

  auto consume = [&](int cur) {
    Frag32 af[2], bf4[4];
#pragma unroll
    for (int f = 0; f < 2; ++f) {
      // A-frag (16x32): lane m=L%16 holds K chunks [8h..+8) and [16+8h..+8)
      const u16* ap = &shA[cur][(wr * 32 + f * 16 + ml) * LDSP];
      af[f].lo = *(const uint4*)(ap + 8 * h);
      af[f].hi = *(const uint4*)(ap + 16 + 8 * h);
    }
#pragma unroll
    for (int f = 0; f < 4; ++f) {
      // B-frag (32x16): lane n=L%16 holds K chunks [8h..+8) and [16+8h..+8)
      const u16* bp = &shB[cur][(wc * 64 + f * 16 + ml) * LDSP];
      bf4[f].lo = *(const uint4*)(bp + 8 * h);
      bf4[f].hi = *(const uint4*)(bp + 16 + 8 * h);
    }
#pragma unroll
    for (int i = 0; i < 2; ++i)
#pragma unroll
      for (int j = 0; j < 4; ++j)
        acc[i][j] = __builtin_amdgcn_wmma_f32_16x16x32_bf16(
            false, __builtin_bit_cast(v16bf, af[i]),
            false, __builtin_bit_cast(v16bf, bf4[j]),
            (short)0, acc[i][j], false, false);
  };

  const int T = Kdim / TK;
  // prologue: tile 0 -> buffer 0
  async_load_b128(ldsA[0],      pa);
  async_load_b128(ldsA[0] + 16, pa + 8);
  async_load_b128(ldsB[0],      pb);
  async_load_b128(ldsB[0] + 16, pb + 8);

  // steady state: unconditional issue of tile t+1 while consuming tile t
  for (int t = 0; t < T - 1; ++t) {
    const int cur = t & 1;
    pa += TK;
    pb += TK;
    __syncthreads();  // WAR: all waves done reading buffer (1-cur) from t-1
    async_load_b128(ldsA[1 - cur],      pa);
    async_load_b128(ldsA[1 - cur] + 16, pa + 8);
    async_load_b128(ldsB[1 - cur],      pb);
    async_load_b128(ldsB[1 - cur] + 16, pb + 8);
    WAIT_ASYNCCNT(4);  // async loads complete in order -> tile t landed
    __syncthreads();   // tile t visible to all waves
    consume(cur);
  }

  // tail: consume last tile
  WAIT_ASYNCCNT(0);
  __syncthreads();
  consume((T - 1) & 1);

  // Epilogue. C-frag layout: VGPR r, lane L -> M = r + 8*(L/16), N = L%16.
  const int nl = lane & 15;
#pragma unroll
  for (int i = 0; i < 2; ++i)
#pragma unroll
    for (int j = 0; j < 4; ++j) {
      const int mBase = rowBase + wr * 32 + i * 16 + 8 * h;
      const int nIdx  = colBase + wc * 64 + j * 16 + nl;
      if (OUTMODE == 0) {
        float* C = (float*)Cv + (long)b * strideC;
#pragma unroll
        for (int r = 0; r < 8; ++r)
          C[(long)(mBase + r) * Nout + nIdx] = acc[i][j][r] * scale;
      } else if (OUTMODE == 1) {
        u16* C = (u16*)Cv + (long)b * strideC;
#pragma unroll
        for (int r = 0; r < 8; ++r)
          C[(long)(mBase + r) * Nout + nIdx] = f32_to_bf16_rne(acc[i][j][r] * scale);
      } else {
        // transposed bf16 store: 8 consecutive M elements -> one b128 store
        u16* C = (u16*)Cv + (long)b * strideC + (long)nIdx * ldcT + mBase;
        uint32_t pk[4];
#pragma unroll
        for (int p = 0; p < 4; ++p) {
          uint32_t lo = f32_to_bf16_rne(acc[i][j][2 * p] * scale);
          uint32_t hi = f32_to_bf16_rne(acc[i][j][2 * p + 1] * scale);
          pk[p] = lo | (hi << 16);
        }
        *(uint4*)C = *(uint4*)pk;
      }
    }
}

// ---------------------------------------------------------------------------
// Elementwise / prep kernels
// ---------------------------------------------------------------------------
__global__ __launch_bounds__(256)
void convert_f32_bf16(const float* __restrict__ in, u16* __restrict__ out, long n4) {
  long i = (long)blockIdx.x * 256 + threadIdx.x;
  if (i >= n4) return;
  float4 v = ((const float4*)in)[i];
  uint2 o;
  o.x = (uint32_t)f32_to_bf16_rne(v.x) | ((uint32_t)f32_to_bf16_rne(v.y) << 16);
  o.y = (uint32_t)f32_to_bf16_rne(v.z) | ((uint32_t)f32_to_bf16_rne(v.w) << 16);
  ((uint2*)out)[i] = o;
}

__global__ __launch_bounds__(256)
void transpose_f32_bf16(const float* __restrict__ W, u16* __restrict__ Wt) {
  __shared__ u16 tile[32][33];
  const int x = blockIdx.x * 32 + threadIdx.x;
  const int y0 = blockIdx.y * 32;
  for (int i = threadIdx.y; i < 32; i += 8)
    tile[i][threadIdx.x] = f32_to_bf16_rne(W[(long)(y0 + i) * DD + x]);
  __syncthreads();
  const int xo = blockIdx.y * 32 + threadIdx.x;
  const int yo0 = blockIdx.x * 32;
  for (int i = threadIdx.y; i < 32; i += 8)
    Wt[(long)(yo0 + i) * DD + xo] = tile[threadIdx.x][i];
}

// one wave32 per row: sigma = 3^(sigmoid(5 * x.Ws) + 1e-5) - 1
__global__ __launch_bounds__(256)
void sigma_kernel(const float* __restrict__ x, const float* __restrict__ Ws,
                  float* __restrict__ sigma) {
  const int row  = blockIdx.x * 8 + (threadIdx.x >> 5);
  const int lane = threadIdx.x & 31;
  const float* xr = x + (long)row * DD;
  float s = 0.f;
  for (int k = lane * 4; k < DD; k += 128) {
    float4 xv = *(const float4*)(xr + k);
    float4 wv = *(const float4*)(Ws + k);
    s += xv.x * wv.x + xv.y * wv.y + xv.z * wv.z + xv.w * wv.w;
  }
  for (int off = 16; off; off >>= 1) s += __shfl_down(s, off, 32);
  if (lane == 0) {
    float sg = 1.f / (1.f + __expf(-5.f * s)) + 1e-5f;
    sg = __expf(sg * 1.0986122886681098f) - 1.f;  // 3^sg - 1
    sigma[row] = sg;
  }
}

__global__ __launch_bounds__(256)
void gaussian_kernel(const float* __restrict__ sigma, float* __restrict__ P,
                     float* __restrict__ sums) {
  const int b = blockIdx.z, i = blockIdx.y;
  const int j = blockIdx.x * 256 + threadIdx.x;
  const float sg = sigma[b * NN + i];
  const float d  = fabsf((float)(i - j));
  const float t  = d / sg;
  const float g  = (0.3989422804014327f / sg) * __expf(-0.5f * t * t);
  P[((long)b * NN + i) * NN + j] = g;

  __shared__ float red[8];
  float w = g;
  for (int off = 16; off; off >>= 1) w += __shfl_down(w, off, 32);
  const int lane = threadIdx.x & 31, wv = threadIdx.x >> 5;
  if (lane == 0) red[wv] = w;
  __syncthreads();
  if (wv == 0) {
    float v = lane < 8 ? red[lane] : 0.f;
    for (int off = 4; off; off >>= 1) v += __shfl_down(v, off, 32);
    if (lane == 0) atomicAdd(&sums[b], v);
  }
}

__global__ __launch_bounds__(256)
void normalize_P(float* __restrict__ P, const float* __restrict__ sums) {
  const int b = blockIdx.y;
  const long idx = (long)blockIdx.x * 256 + threadIdx.x;
  P[(long)b * NN * NN + idx] *= (1.f / sums[b]);
}

// softmax over the BATCH dim (8 strided values per (n,m)); emit bf16 S
__global__ __launch_bounds__(256)
void softmax_batch_kernel(const float* __restrict__ scores, u16* __restrict__ S) {
  const long idx = (long)blockIdx.x * 256 + threadIdx.x;  // 0 .. N*N-1
  const long st = (long)NN * NN;
  float v[BB], mx = -3.4e38f;
#pragma unroll
  for (int b = 0; b < BB; ++b) { v[b] = scores[b * st + idx]; mx = fmaxf(mx, v[b]); }
  float s = 0.f;
#pragma unroll
  for (int b = 0; b < BB; ++b) { v[b] = __expf(v[b] - mx); s += v[b]; }
  const float r = 1.f / s;
#pragma unroll
  for (int b = 0; b < BB; ++b) S[b * st + idx] = f32_to_bf16_rne(v[b] * r);
}

// ---------------------------------------------------------------------------
extern "C" void kernel_launch(void* const* d_in, const int* in_sizes, int n_in,
                              void* d_out, int out_size, void* d_ws, size_t ws_size,
                              hipStream_t stream) {
  (void)in_sizes; (void)n_in; (void)out_size; (void)ws_size;
  const float* x  = (const float*)d_in[0];
  const float* Wq = (const float*)d_in[1];
  const float* Wk = (const float*)d_in[2];
  const float* Wv = (const float*)d_in[3];
  const float* Ws = (const float*)d_in[4];
  float* Z = (float*)d_out;                            // [B,N,D]
  float* P = (float*)d_out + (size_t)BB * NN * DD;     // [B,N,N]

  char* ws = (char*)d_ws;
  size_t off = 0;
  auto carve = [&](size_t bytes) -> char* {
    char* p = ws + off;
    off = (off + bytes + 255) & ~(size_t)255;
    return p;
  };
  u16*   x_bf   = (u16*)carve((size_t)BB * NN * DD * 2);
  u16*   Wqt    = (u16*)carve((size_t)DD * DD * 2);
  u16*   Wkt    = (u16*)carve((size_t)DD * DD * 2);
  u16*   Wvt    = (u16*)carve((size_t)DD * DD * 2);
  u16*   Qbf    = (u16*)carve((size_t)BB * NN * DD * 2);
  u16*   Kbf    = (u16*)carve((size_t)BB * NN * DD * 2);
  u16*   Vt     = (u16*)carve((size_t)BB * DD * NN * 2);   // V^T per batch
  u16*   Sbf    = (u16*)carve((size_t)BB * NN * NN * 2);
  float* scores = (float*)carve((size_t)BB * NN * NN * 4);
  float* sigma  = (float*)carve((size_t)BB * NN * 4);
  float* sums   = (float*)carve((size_t)BB * 4);

  // precision conversions
  convert_f32_bf16<<<(BB * NN * DD / 4 + 255) / 256, 256, 0, stream>>>(
      x, x_bf, (long)BB * NN * DD / 4);
  dim3 tb(32, 8), tg(DD / 32, DD / 32);
  transpose_f32_bf16<<<tg, tb, 0, stream>>>(Wq, Wqt);
  transpose_f32_bf16<<<tg, tb, 0, stream>>>(Wk, Wkt);
  transpose_f32_bf16<<<tg, tb, 0, stream>>>(Wv, Wvt);

  // prior association P
  sigma_kernel<<<BB * NN / 8, 256, 0, stream>>>(x, Ws, sigma);
  hipMemsetAsync(sums, 0, BB * sizeof(float), stream);
  gaussian_kernel<<<dim3(NN / 256, NN, BB), 256, 0, stream>>>(sigma, P, sums);
  normalize_P<<<dim3(NN * NN / 256, BB), 256, 0, stream>>>(P, sums);

  // projections (bf16 WMMA); V written transposed for the Z GEMM
  dim3 blk(256), pg(DD / TN, NN / TM, BB);
  gemm_bf16_wmma<1><<<pg, blk, 0, stream>>>(x_bf, Wqt, Qbf, NN, DD, DD,
      (long)NN * DD, 0L, (long)NN * DD, 1.f, 0);
  gemm_bf16_wmma<1><<<pg, blk, 0, stream>>>(x_bf, Wkt, Kbf, NN, DD, DD,
      (long)NN * DD, 0L, (long)NN * DD, 1.f, 0);
  gemm_bf16_wmma<2><<<pg, blk, 0, stream>>>(x_bf, Wvt, Vt, NN, DD, DD,
      (long)NN * DD, 0L, (long)DD * NN, 1.f, NN);

  // scores = Q K^T / sqrt(D); rows of K ARE the Bt layout
  dim3 sg(NN / TN, NN / TM, BB);
  gemm_bf16_wmma<0><<<sg, blk, 0, stream>>>(Qbf, Kbf, scores, NN, NN, DD,
      (long)NN * DD, (long)NN * DD, (long)NN * NN, 0.03125f, 0);

  // softmax over batch dim -> bf16 S
  softmax_batch_kernel<<<NN * NN / 256, 256, 0, stream>>>(scores, Sbf);

  // Z = S V  (Bt = V^T)
  dim3 zg(DD / TN, NN / TM, BB);
  gemm_bf16_wmma<0><<<zg, blk, 0, stream>>>(Sbf, Vt, Z, NN, DD, NN,
      (long)NN * NN, (long)DD * NN, (long)NN * DD, 1.f, 0);
}
